// Recommender_49563922596401
// MI455X (gfx1250) — matrix operand order repaired and checked by reading, not compiled, not run on hardware
//
#include <hip/hip_runtime.h>

#define N_ENT 200000
#define N_USR 100000
#define DIM   64
#define NEDGE 1500000
#define NINT  1000000
#define EPSF  1e-12f

typedef __attribute__((ext_vector_type(16))) _Float16 v16h;
typedef __attribute__((ext_vector_type(8)))  float    v8f;

__device__ __forceinline__ float wsum32(float v) {
  v += __shfl_xor(v, 16, 32);
  v += __shfl_xor(v, 8, 32);
  v += __shfl_xor(v, 4, 32);
  v += __shfl_xor(v, 2, 32);
  v += __shfl_xor(v, 1, 32);
  return v;
}

__device__ __forceinline__ void atomAdd(float* p, float v) {
  (void)__hip_atomic_fetch_add(p, v, __ATOMIC_RELAXED, __HIP_MEMORY_SCOPE_AGENT);
}

// ---------------------------------------------------------------------------
// Relation remap: scores = rel_weight[10,64] @ latent[3,64]^T via two chained
// v_wmma_f32_16x16x32_f16 (K = 64 = 2*32). One wave. Rows padded to 16,
// columns (virtual relations) padded to 16. Loads are unconditional from
// clamped row indices; padding rows/cols are zeroed by a multiplicative mask
// so control flow stays uniform (no exec-mask churn around each load).
// A 16x32 f16 layout: lane L (M=L&15): elems e<8 -> K = base + e (+8 if L>=16),
//                     elems e>=8   -> K = base + e+8 (+8 if L>=16)
// B 32x16 f16 layout: lane L (N=L&15): same K mapping as A
// C 16x16 f32: lane L (N=L&15), vgpr r -> M = r (+8 if L>=16)
// ---------------------------------------------------------------------------
__global__ __launch_bounds__(32) void k_remap(const float* __restrict__ relw,
                                              const float* __restrict__ lat,
                                              int* __restrict__ remap) {
  const int lane = threadIdx.x;
  const int m = lane & 15;
  const bool hi = lane >= 16;
  const float amask = (m < 10) ? 1.0f : 0.0f;
  const float bmask = (m < 3) ? 1.0f : 0.0f;
  const float* arow = relw + (size_t)min(m, 9) * 64;
  const float* brow = lat + (size_t)min(m, 2) * 64;
  v8f c = {};
  #pragma unroll
  for (int kb = 0; kb < 64; kb += 32) {
    v16h a, b;
    #pragma unroll
    for (int e = 0; e < 16; ++e) {
      int ka = kb + (hi ? 8 : 0) + (e < 8 ? e : e + 8);
      a[e] = (_Float16)(arow[ka] * amask);
      int kbn = kb + (hi ? 8 : 0) + (e < 8 ? e : e + 8);
      b[e] = (_Float16)(brow[kbn] * bmask);
    }
    c = __builtin_amdgcn_wmma_f32_16x16x32_f16(false, a, false, b, (short)0, c,
                                               false, false);
  }
  __shared__ float sc[16][16];
  #pragma unroll
  for (int r = 0; r < 8; ++r) sc[hi ? (8 + r) : r][m] = c[r];
  __syncthreads();
  if (lane < 10) {
    float b0 = sc[lane][0], b1 = sc[lane][1], b2 = sc[lane][2];
    int best = 0; float bv = b0;
    if (b1 > bv) { bv = b1; best = 1; }
    if (b2 > bv) { bv = b2; best = 2; }
    remap[lane] = best;
  }
}

// ---------------------------------------------------------------------------
__global__ __launch_bounds__(256) void k_zero(float* __restrict__ p, int n4) {
  int i = blockIdx.x * blockDim.x + threadIdx.x;
  int stride = gridDim.x * blockDim.x;
  float4 z = make_float4(0.f, 0.f, 0.f, 0.f);
  for (; i < n4; i += stride) ((float4*)p)[i] = z;
}

__global__ __launch_bounds__(256) void k_et_cnt(const int* __restrict__ etype,
                                                const int* __restrict__ head,
                                                const int* __restrict__ remap,
                                                int* __restrict__ et,
                                                float* __restrict__ cntE) {
  int e = blockIdx.x * blockDim.x + threadIdx.x;
  if (e >= NEDGE) return;
  int v = remap[etype[e] - 1];
  et[e] = v;
  atomAdd(&cntE[(size_t)v * N_ENT + head[e]], 1.0f);
}

__global__ __launch_bounds__(256) void k_cnt_usr(const int* __restrict__ uidx,
                                                 float* __restrict__ cntU) {
  int i = blockIdx.x * blockDim.x + threadIdx.x;
  if (i >= NINT) return;
  atomAdd(&cntU[uidx[i]], 1.0f);
}

// ---------------------------------------------------------------------------
// Entity LWS: per-edge cumulative scale s; neigh_cur = s * ent[tail]
// sim pass:   db = dot(u_v[head], ent[tail]);  s <- s*s*db
// scatter:    acc_v[head] += s * ent[tail]     (s==1 at it==0)
// ---------------------------------------------------------------------------
__global__ __launch_bounds__(256) void k_edge_sim(const int* __restrict__ head,
                                                  const int* __restrict__ tail,
                                                  const int* __restrict__ et,
                                                  const float* __restrict__ ent,
                                                  const float* __restrict__ uvb,
                                                  float* __restrict__ s) {
  int gid = blockIdx.x * blockDim.x + threadIdx.x;
  int e = gid >> 5, lane = gid & 31;
  if (e >= NEDGE) return;
  int h = head[e], t = tail[e], v = et[e];
  const float2* up = (const float2*)(uvb + ((size_t)v * N_ENT + h) * DIM);
  const float2* ep = (const float2*)(ent + (size_t)t * DIM);
  float2 a = up[lane], b = ep[lane];
  float db = wsum32(a.x * b.x + a.y * b.y);
  if (lane == 0) { float sv = s[e]; s[e] = sv * sv * db; }
}

__global__ __launch_bounds__(256) void k_edge_scatter(const int* __restrict__ head,
                                                      const int* __restrict__ tail,
                                                      const int* __restrict__ et,
                                                      const float* __restrict__ ent,
                                                      float* __restrict__ uvb,
                                                      float* __restrict__ s,
                                                      int it) {
  int gid = blockIdx.x * blockDim.x + threadIdx.x;
  int e = gid >> 5, lane = gid & 31;
  if (e >= NEDGE) return;
  int h = head[e], t = tail[e], v = et[e];
  float scale;
  if (it == 0) { scale = 1.0f; if (lane == 0) s[e] = 1.0f; }
  else         { scale = s[e]; }
  float2 b = ((const float2*)(ent + (size_t)t * DIM))[lane];
  float* dst = uvb + ((size_t)v * N_ENT + h) * DIM + lane * 2;
  atomAdd(dst,     scale * b.x);
  atomAdd(dst + 1, scale * b.y);
}

// acc -> u in place:  u = acc/denom; squash (it<2); u += ent
__global__ __launch_bounds__(256) void k_node_ent(float* __restrict__ uvb,
                                                  const float* __restrict__ ent,
                                                  const float* __restrict__ cntE,
                                                  int it) {
  int gid = blockIdx.x * blockDim.x + threadIdx.x;
  int w = gid >> 5, lane = gid & 31;
  if (w >= 3 * N_ENT) return;
  int v = w / N_ENT, n = w - v * N_ENT;
  float rden = 1.0f / fmaxf(cntE[(size_t)v * N_ENT + n], 1.0f);
  float2* p = (float2*)(uvb + ((size_t)v * N_ENT + n) * DIM);
  float2 a = p[lane];
  a.x *= rden; a.y *= rden;
  float sq = wsum32(a.x * a.x + a.y * a.y);
  if (it < 2) {
    float nrm = sqrtf(sq);
    float scl = sq / ((sq + 1.0f) * fmaxf(nrm, EPSF));
    a.x *= scl; a.y *= scl;
  }
  float2 eb = ((const float2*)(ent + (size_t)n * DIM))[lane];
  a.x += eb.x; a.y += eb.y;
  p[lane] = a;
}

// ---------------------------------------------------------------------------
// User LWS: neigh re-gathered fresh each iteration -> plain per-iter sim
// ---------------------------------------------------------------------------
__global__ __launch_bounds__(256) void k_user_sim(const int* __restrict__ uidx,
                                                  const int* __restrict__ iidx,
                                                  const float* __restrict__ ent,
                                                  const float* __restrict__ uub,
                                                  float* __restrict__ simu) {
  int gid = blockIdx.x * blockDim.x + threadIdx.x;
  int i = gid >> 5, lane = gid & 31;
  if (i >= NINT) return;
  int u = uidx[i], t = iidx[i];
  float2 a = ((const float2*)(uub + (size_t)u * DIM))[lane];
  float2 b = ((const float2*)(ent + (size_t)t * DIM))[lane];
  float db = wsum32(a.x * b.x + a.y * b.y);
  if (lane == 0) simu[i] = db;
}

__global__ __launch_bounds__(256) void k_user_scatter(const int* __restrict__ uidx,
                                                      const int* __restrict__ iidx,
                                                      const float* __restrict__ ent,
                                                      float* __restrict__ uub,
                                                      const float* __restrict__ simu,
                                                      int it) {
  int gid = blockIdx.x * blockDim.x + threadIdx.x;
  int i = gid >> 5, lane = gid & 31;
  if (i >= NINT) return;
  int u = uidx[i], t = iidx[i];
  float scale = (it == 0) ? 1.0f : simu[i];
  float2 b = ((const float2*)(ent + (size_t)t * DIM))[lane];
  float* dst = uub + (size_t)u * DIM + lane * 2;
  atomAdd(dst,     scale * b.x);
  atomAdd(dst + 1, scale * b.y);
}

__global__ __launch_bounds__(256) void k_node_usr(float* __restrict__ uub,
                                                  const float* __restrict__ usr,
                                                  const float* __restrict__ cntU,
                                                  int it) {
  int gid = blockIdx.x * blockDim.x + threadIdx.x;
  int n = gid >> 5, lane = gid & 31;
  if (n >= N_USR) return;
  float rden = 1.0f / fmaxf(cntU[n], 1.0f);
  float2* p = (float2*)(uub + (size_t)n * DIM);
  float2 a = p[lane];
  a.x *= rden; a.y *= rden;
  float sq = wsum32(a.x * a.x + a.y * a.y);
  if (it < 2) {
    float nrm = sqrtf(sq);
    float scl = sq / ((sq + 1.0f) * fmaxf(nrm, EPSF));
    a.x *= scl; a.y *= scl;
  }
  float2 ub = ((const float2*)(usr + (size_t)n * DIM))[lane];
  a.x += ub.x; a.y += ub.y;
  p[lane] = a;
}

// ---------------------------------------------------------------------------
// Per-hop combine: ent <- l2norm(softmax(agg_w[hop]) . u_v); out += ent
// ---------------------------------------------------------------------------
__global__ __launch_bounds__(256) void k_combine_ent(const float* __restrict__ uvb,
                                                     const float* __restrict__ aggw,
                                                     float* __restrict__ entC,
                                                     float* __restrict__ outE,
                                                     int hop) {
  int gid = blockIdx.x * blockDim.x + threadIdx.x;
  int n = gid >> 5, lane = gid & 31;
  if (n >= N_ENT) return;
  float a0 = aggw[hop * 3 + 0], a1 = aggw[hop * 3 + 1], a2 = aggw[hop * 3 + 2];
  float mx = fmaxf(a0, fmaxf(a1, a2));
  float e0 = __expf(a0 - mx), e1 = __expf(a1 - mx), e2 = __expf(a2 - mx);
  float inv = 1.0f / (e0 + e1 + e2);
  float w0 = e0 * inv, w1 = e1 * inv, w2 = e2 * inv;
  float2 x0 = ((const float2*)(uvb + ((size_t)0 * N_ENT + n) * DIM))[lane];
  float2 x1 = ((const float2*)(uvb + ((size_t)1 * N_ENT + n) * DIM))[lane];
  float2 x2 = ((const float2*)(uvb + ((size_t)2 * N_ENT + n) * DIM))[lane];
  float2 g;
  g.x = w0 * x0.x + w1 * x1.x + w2 * x2.x;
  g.y = w0 * x0.y + w1 * x1.y + w2 * x2.y;
  float sq = wsum32(g.x * g.x + g.y * g.y);
  float scl = 1.0f / fmaxf(sqrtf(sq), EPSF);
  g.x *= scl; g.y *= scl;
  ((float2*)(entC + (size_t)n * DIM))[lane] = g;
  float2* o = (float2*)(outE + (size_t)n * DIM);
  float2 ov = o[lane];
  ov.x += g.x; ov.y += g.y;
  o[lane] = ov;
}

__global__ __launch_bounds__(256) void k_combine_usr(const float* __restrict__ uub,
                                                     float* __restrict__ usrC,
                                                     float* __restrict__ outU) {
  int gid = blockIdx.x * blockDim.x + threadIdx.x;
  int n = gid >> 5, lane = gid & 31;
  if (n >= N_USR) return;
  float2 g = ((const float2*)(uub + (size_t)n * DIM))[lane];
  float sq = wsum32(g.x * g.x + g.y * g.y);
  float scl = 1.0f / fmaxf(sqrtf(sq), EPSF);
  g.x *= scl; g.y *= scl;
  ((float2*)(usrC + (size_t)n * DIM))[lane] = g;
  float2* o = (float2*)(outU + (size_t)n * DIM);
  float2 ov = o[lane];
  ov.x += g.x; ov.y += g.y;
  o[lane] = ov;
}

// ---------------------------------------------------------------------------
extern "C" void kernel_launch(void* const* d_in, const int* in_sizes, int n_in,
                              void* d_out, int out_size, void* d_ws, size_t ws_size,
                              hipStream_t stream) {
  const float* entity_emb = (const float*)d_in[0];
  const float* user_emb   = (const float*)d_in[1];
  const float* latent     = (const float*)d_in[2];
  const float* relw       = (const float*)d_in[3];
  const float* aggw       = (const float*)d_in[4];
  const int*   eidx       = (const int*)d_in[5];
  const int*   etype      = (const int*)d_in[6];
  const int*   uidx       = (const int*)d_in[7];
  const int*   iidx       = (const int*)d_in[8];
  const int* head = eidx;
  const int* tail = eidx + NEDGE;

  // workspace carve (~275 MB total)
  char* w = (char*)d_ws;
  auto take = [&](size_t bytes) -> char* {
    char* p = w;
    w += (bytes + 255) & ~(size_t)255;
    return p;
  };
  int*   remap = (int*)  take(64);
  int*   et    = (int*)  take((size_t)NEDGE * 4);
  float* sbuf  = (float*)take((size_t)NEDGE * 4);
  float* simu  = (float*)take((size_t)NINT * 4);
  float* cntE  = (float*)take((size_t)3 * N_ENT * 4);
  float* cntU  = (float*)take((size_t)N_USR * 4);
  float* entC  = (float*)take((size_t)N_ENT * DIM * 4);
  float* usrC  = (float*)take((size_t)N_USR * DIM * 4);
  float* uvb   = (float*)take((size_t)3 * N_ENT * DIM * 4);
  float* uub   = (float*)take((size_t)N_USR * DIM * 4);

  float* out = (float*)d_out;
  hipMemcpyAsync(entC, entity_emb, (size_t)N_ENT * DIM * 4, hipMemcpyDeviceToDevice, stream);
  hipMemcpyAsync(usrC, user_emb,   (size_t)N_USR * DIM * 4, hipMemcpyDeviceToDevice, stream);
  hipMemcpyAsync(out,  entity_emb, (size_t)N_ENT * DIM * 4, hipMemcpyDeviceToDevice, stream);
  hipMemcpyAsync(out + (size_t)N_ENT * DIM, user_emb, (size_t)N_USR * DIM * 4,
                 hipMemcpyDeviceToDevice, stream);

  k_remap<<<1, 32, 0, stream>>>(relw, latent, remap);
  k_zero<<<1024, 256, 0, stream>>>(cntE, (3 * N_ENT) / 4);
  k_zero<<<256, 256, 0, stream>>>(cntU, N_USR / 4);
  k_et_cnt<<<(NEDGE + 255) / 256, 256, 0, stream>>>(etype, head, remap, et, cntE);
  k_cnt_usr<<<(NINT + 255) / 256, 256, 0, stream>>>(uidx, cntU);

  const int EB   = (NEDGE * 32) / 256;      // 187500 edge-waves blocks
  const int NB3  = (3 * N_ENT * 32) / 256;  // 75000
  const int UB   = (NINT * 32) / 256;       // 125000
  const int USRB = (N_USR * 32) / 256;      // 12500
  const int ENTB = (N_ENT * 32) / 256;      // 25000

  for (int hop = 0; hop < 2; ++hop) {
    // entity LWS over all 3 virtual relations simultaneously
    for (int it = 0; it < 3; ++it) {
      if (it > 0) k_edge_sim<<<EB, 256, 0, stream>>>(head, tail, et, entC, uvb, sbuf);
      k_zero<<<8192, 256, 0, stream>>>(uvb, (3 * N_ENT * DIM) / 4);
      k_edge_scatter<<<EB, 256, 0, stream>>>(head, tail, et, entC, uvb, sbuf, it);
      k_node_ent<<<NB3, 256, 0, stream>>>(uvb, entC, cntE, it);
    }
    // user LWS (uses the pre-update entC)
    for (int it = 0; it < 3; ++it) {
      if (it > 0) k_user_sim<<<UB, 256, 0, stream>>>(uidx, iidx, entC, uub, simu);
      k_zero<<<4096, 256, 0, stream>>>(uub, (N_USR * DIM) / 4);
      k_user_scatter<<<UB, 256, 0, stream>>>(uidx, iidx, entC, uub, simu, it);
      k_node_usr<<<USRB, 256, 0, stream>>>(uub, usrC, cntU, it);
    }
    k_combine_ent<<<ENTB, 256, 0, stream>>>(uvb, aggw, entC, out, hop);
    k_combine_usr<<<USRB, 256, 0, stream>>>(uub, usrC, out + (size_t)N_ENT * DIM);
  }
}
